// GraphGPS_3453153706292
// MI455X (gfx1250) — matrix operand order repaired and checked by reading, not compiled, not run on hardware
//
#include <hip/hip_runtime.h>

#define NN 4096
#define EE 65536
#define GG 64
#define HH 128
#define HEADS 4
#define HD 32
#define COUT 10

typedef __attribute__((ext_vector_type(16))) __bf16    v16bf;
typedef __attribute__((ext_vector_type(8)))  float     v8f;
typedef __attribute__((ext_vector_type(8)))  unsigned  v8u;

// ---------------------------------------------------------------------------
// WMMA helpers (CDNA5 ISA 7.12.2 VGPR layouts, bf16 operands, f32 accum)
// ---------------------------------------------------------------------------
__device__ __forceinline__ v8f wmma_bf16(v16bf a, v16bf b, v8f c) {
  return __builtin_amdgcn_wmma_f32_16x16x32_bf16(false, a, false, b, (short)0, c,
                                                 false, false);
}

__device__ __forceinline__ unsigned pack_bf16(float lo, float hi) {
  union { __bf16 h[2]; unsigned u; } t;
  t.h[0] = (__bf16)lo;
  t.h[1] = (__bf16)hi;
  return t.u;
}

// A fragment: 16x32 tile of row-major f32 matrix A (lda = row stride).
// lanes 0-15: M=lane, K 0-15; lanes 16-31: M=lane-16, K 16-31
__device__ __forceinline__ v16bf wmma_a_f32(const float* A, int lda, int row0, int k0) {
  int lane = threadIdx.x & 31;
  int row = row0 + (lane & 15);
  int khalf = lane >> 4;
  v16bf a;
#pragma unroll
  for (int r = 0; r < 8; ++r) {
    int kb = k0 + (khalf << 4) + (r << 1);
    const float* p = A + (size_t)row * lda + kb;
    a[2 * r]     = (__bf16)p[0];
    a[2 * r + 1] = (__bf16)p[1];
  }
  return a;
}

// A fragment straight from bf16 memory: K-pairs (2m,2m+1) are contiguous ->
// one dword load per pair, zero conversion VALU.
__device__ __forceinline__ v16bf wmma_a_bf16(const __bf16* A, int lda, int row0) {
  int lane = threadIdx.x & 31;
  int row = row0 + (lane & 15);
  int khalf = lane >> 4;
  union { v8u u; v16bf b; } t;
#pragma unroll
  for (int r = 0; r < 8; ++r) {
    int kb = (khalf << 4) + (r << 1);
    t.u[r] = *(const unsigned*)(A + (size_t)row * lda + kb);  // kb even -> 4B aligned
  }
  return t.b;
}

// A fragment from f32 LDS staging tile
__device__ __forceinline__ v16bf wmma_a_lds(const float* S, int ld) {
  int lane = threadIdx.x & 31;
  const float* base = S + (lane & 15) * ld;
  int khalf = lane >> 4;
  v16bf a;
#pragma unroll
  for (int r = 0; r < 8; ++r) {
    int kb = (khalf << 4) + (r << 1);
    a[2 * r]     = (__bf16)base[kb];
    a[2 * r + 1] = (__bf16)base[kb + 1];
  }
  return a;
}

// B fragment from pre-packed bf16 weights: P[m*Nc + c] packs rows (2m,2m+1).
// 8 coalesced dword loads per lane, zero conversion VALU.
__device__ __forceinline__ v16bf wmma_b_packed(const unsigned* P, int Nc, int k0,
                                               int col0) {
  int lane = threadIdx.x & 31;
  int col = col0 + (lane & 15);
  int mb = (k0 >> 1) + ((lane >> 4) << 3);  // k0/2 + khalf*8
  union { v8u u; v16bf b; } t;
#pragma unroll
  for (int r = 0; r < 8; ++r) t.u[r] = P[(size_t)(mb + r) * Nc + col];
  return t.b;
}

// B fragment from bf16 row-major [K x N]: pack two 16-bit loads per pair.
__device__ __forceinline__ v16bf wmma_b_bf16(const __bf16* B, int ldb, int k0,
                                             int col0) {
  int lane = threadIdx.x & 31;
  int col = col0 + (lane & 15);
  int kb = k0 + ((lane >> 4) << 4);
  union { v8u u; v16bf b; } t;
#pragma unroll
  for (int r = 0; r < 8; ++r) {
    unsigned lo = *(const unsigned short*)(B + (size_t)(kb + 2 * r) * ldb + col);
    unsigned hi = *(const unsigned short*)(B + (size_t)(kb + 2 * r + 1) * ldb + col);
    t.u[r] = lo | (hi << 16);
  }
  return t.b;
}

// Transposed B fragment from bf16 memory: logical B(k,n) = Bt[(col0+n)*ld + k].
// Each lane reads 16 consecutive bf16 -> 8 raw dword loads, zero VALU.
__device__ __forceinline__ v16bf wmma_bT_bf16(const __bf16* Bt, int ld, int col0) {
  int lane = threadIdx.x & 31;
  const __bf16* p = Bt + (size_t)(col0 + (lane & 15)) * ld + ((lane >> 4) << 4);
  const unsigned* q = (const unsigned*)p;  // 32B-aligned offsets throughout
  union { v8u u; v16bf b; } t;
#pragma unroll
  for (int r = 0; r < 8; ++r) t.u[r] = q[r];
  return t.b;
}

// ---------------------------------------------------------------------------
// DPP-based 16-lane-group reductions (no LDS, no ds_bpermute)
// ---------------------------------------------------------------------------
template <int CTRL>
__device__ __forceinline__ float dpp_xf(float v) {
  int r = __builtin_amdgcn_update_dpp(0, __float_as_int(v), CTRL, 0xF, 0xF, true);
  return __int_as_float(r);
}

__device__ __forceinline__ float row16_max(float v) {
  v = fmaxf(v, dpp_xf<0xB1>(v));   // quad_perm {1,0,3,2}  (xor 1)
  v = fmaxf(v, dpp_xf<0x4E>(v));   // quad_perm {2,3,0,1}  (xor 2)
  v = fmaxf(v, dpp_xf<0x141>(v));  // row_half_mirror      (xor within 8)
  v = fmaxf(v, dpp_xf<0x140>(v));  // row_mirror           (xor within 16)
  return v;
}

__device__ __forceinline__ float row16_sum(float v) {
  v += dpp_xf<0xB1>(v);
  v += dpp_xf<0x4E>(v);
  v += dpp_xf<0x141>(v);
  v += dpp_xf<0x140>(v);
  return v;
}

// ---------------------------------------------------------------------------
// Generic WMMA GEMM: C[M,Nc] = act(A[M,K] @ W + bias (+ res)); W pre-packed bf16.
// One wave -> 16x64 output tile. K mult of 32, Nc mult of 64, M mult of 16.
// ---------------------------------------------------------------------------
template <bool RELU, bool RES, bool OUT16>
__global__ void __launch_bounds__(128) gemm_bf16_kernel(
    const float* __restrict__ A, const unsigned* __restrict__ P,
    const float* __restrict__ bias, const float* __restrict__ res,
    void* __restrict__ Cv, int M, int Nc, int K) {
  int wid = (int)((blockIdx.x * blockDim.x + threadIdx.x) >> 5);
  int ntiles = Nc >> 6;
  if (wid >= (M >> 4) * ntiles) return;
  int tm = wid / ntiles, tn = wid - tm * ntiles;
  int row0 = tm << 4, col0 = tn << 6;

  v8f acc[4] = {};
  for (int k0 = 0; k0 < K; k0 += 32) {
    v16bf a = wmma_a_f32(A, K, row0, k0);
#pragma unroll
    for (int j = 0; j < 4; ++j) {
      v16bf b = wmma_b_packed(P, Nc, k0, col0 + j * 16);
      acc[j] = wmma_bf16(a, b, acc[j]);
    }
  }

  int lane = threadIdx.x & 31;
  int hi = lane >> 4, nl = lane & 15;
#pragma unroll
  for (int j = 0; j < 4; ++j) {
    int col = col0 + j * 16 + nl;
    float bv = bias[col];
#pragma unroll
    for (int r = 0; r < 8; ++r) {
      int row = row0 + r + (hi << 3);
      float v = acc[j][r] + bv;
      if (RES) v += res[(size_t)row * Nc + col];
      if (RELU) v = fmaxf(v, 0.0f);
      if (OUT16)
        ((__bf16*)Cv)[(size_t)row * Nc + col] = (__bf16)v;
      else
        ((float*)Cv)[(size_t)row * Nc + col] = v;
    }
  }
}

// ---------------------------------------------------------------------------
// Fused flash attention over bf16 qkv [NN, 3*HH]; 4 heads, head_dim 32.
// One wave per (head, 16-query tile); online softmax state in registers.
// ---------------------------------------------------------------------------
__global__ void __launch_bounds__(128) attn_kernel(const __bf16* __restrict__ qkv,
                                                   float* __restrict__ ob) {
  __shared__ float pbuf[4][16][34];
  int lw = threadIdx.x >> 5;
  int lane = threadIdx.x & 31;
  int wid = blockIdx.x * 4 + lw;
  int head = wid >> 8;  // NN/16 = 256 query tiles per head
  int qt = wid & 255;
  if (head >= HEADS) return;
  int q0 = qt << 4;

  const __bf16* Qp = qkv + head * HD;
  const __bf16* Kp = qkv + HH + head * HD;
  const __bf16* Vp = qkv + 2 * HH + head * HD;
  const float scale = 0.17677669529663687f;  // 1/sqrt(32), applied post-WMMA

  v16bf qa = wmma_a_bf16(Qp, 3 * HH, q0);
  v8f o0 = {}, o1 = {};
  float mrow[8], lrow[8];
#pragma unroll
  for (int r = 0; r < 8; ++r) { mrow[r] = -3.0e38f; lrow[r] = 0.0f; }
  int hi = lane >> 4, nl = lane & 15;
  float* prow = &pbuf[lw][0][0];

  for (int c0 = 0; c0 < NN; c0 += 32) {
    v16bf kb0 = wmma_bT_bf16(Kp, 3 * HH, c0);
    v16bf kb1 = wmma_bT_bf16(Kp, 3 * HH, c0 + 16);
    v8f s0 = {}, s1 = {};
    s0 = wmma_bf16(qa, kb0, s0);  // S[16q x 16k], keys c0..c0+15
    s1 = wmma_bf16(qa, kb1, s1);  // keys c0+16..c0+31

    // online softmax; C-layout: VGPR r -> row (r + 8*hi); lanes of a 16-group
    // share a row, so DPP xor-style reductions stay in-row.
#pragma unroll
    for (int r = 0; r < 8; ++r) {
      float a0 = s0[r] * scale;
      float a1 = s1[r] * scale;
      float mx = row16_max(fmaxf(a0, a1));
      float mnew = fmaxf(mrow[r], mx);
      float corr = __expf(mrow[r] - mnew);
      float e0 = __expf(a0 - mnew);
      float e1 = __expf(a1 - mnew);
      float rs = row16_sum(e0 + e1);
      lrow[r] = lrow[r] * corr + rs;
      mrow[r] = mnew;
      o0[r] *= corr;
      o1[r] *= corr;
      prow[(r + (hi << 3)) * 34 + nl] = e0;        // P tile -> LDS (C-layout)
      prow[(r + (hi << 3)) * 34 + 16 + nl] = e1;
    }
    // per-wave LDS is in-order; fence only stops compiler reordering
    asm volatile("" ::: "memory");
    v16bf pa = wmma_a_lds(prow, 34);               // reload as A fragment
    asm volatile("" ::: "memory");
    v16bf vb0 = wmma_b_bf16(Vp, 3 * HH, c0, 0);
    v16bf vb1 = wmma_b_bf16(Vp, 3 * HH, c0, 16);
    o0 = wmma_bf16(pa, vb0, o0);
    o1 = wmma_bf16(pa, vb1, o1);
  }

#pragma unroll
  for (int r = 0; r < 8; ++r) {
    float inv = 1.0f / lrow[r];
    int row = q0 + r + (hi << 3);
    ob[(size_t)row * HH + head * HD + nl] = o0[r] * inv;
    ob[(size_t)row * HH + head * HD + 16 + nl] = o1[r] * inv;
  }
}

// ---------------------------------------------------------------------------
// Glue kernels
// ---------------------------------------------------------------------------
__global__ void zero_kernel(float* p, int n) {
  int i = blockIdx.x * blockDim.x + threadIdx.x;
  if (i < n) p[i] = 0.0f;
}

__global__ void add_kernel(const float* a, const float* b, float* c, int n) {
  int i = blockIdx.x * blockDim.x + threadIdx.x;
  if (i < n) c[i] = a[i] + b[i];
}

__global__ void build_xin_kernel(const float* __restrict__ x,
                                 const float* __restrict__ lap,
                                 const float* __restrict__ rw,
                                 float* __restrict__ xin) {
  int i = blockIdx.x * blockDim.x + threadIdx.x;
  if (i >= NN * 96) return;
  int n = i / 96, c = i - n * 96;
  float v = 0.0f;
  if (c < 64) v = x[n * 64 + c];
  else if (c < 72) v = lap[n * 8 + (c - 64)];
  else if (c < 88) v = rw[n * 16 + (c - 72)];
  xin[i] = v;
}

// pad K 88->96 and pack bf16 pairs in one pass
__global__ void pack_wp_kernel(const float* __restrict__ Wp, unsigned* __restrict__ P) {
  int i = blockIdx.x * blockDim.x + threadIdx.x;
  if (i >= 48 * HH) return;
  int m = i / HH, c = i - m * HH;
  int k0 = 2 * m, k1 = 2 * m + 1;
  float lo = (k0 < 88) ? Wp[k0 * HH + c] : 0.0f;
  float hi = (k1 < 88) ? Wp[k1 * HH + c] : 0.0f;
  P[i] = pack_bf16(lo, hi);
}

// pack rows (2m,2m+1) of row-major W[K x Nc] into dwords of bf16 pairs
__global__ void pack_weight_kernel(const float* __restrict__ W,
                                   unsigned* __restrict__ P, int K, int Nc) {
  int i = blockIdx.x * blockDim.x + threadIdx.x;
  if (i >= (K >> 1) * Nc) return;
  int m = i / Nc, c = i - m * Nc;
  P[i] = pack_bf16(W[(size_t)(2 * m) * Nc + c], W[(size_t)(2 * m + 1) * Nc + c]);
}

__global__ void scatter_add_kernel(const float* __restrict__ h,
                                   const int* __restrict__ src,
                                   const int* __restrict__ dst,
                                   float* __restrict__ agg) {
  int idx = blockIdx.x * blockDim.x + threadIdx.x;
  if (idx >= EE * 32) return;
  int e = idx >> 5;
  int c = (idx & 31) << 2;
  int s = src[e], d = dst[e];
  const float* hp = h + (size_t)s * HH + c;
  float* ap = agg + (size_t)d * HH + c;
  atomicAdd(ap + 0, hp[0]);
  atomicAdd(ap + 1, hp[1]);
  atomicAdd(ap + 2, hp[2]);
  atomicAdd(ap + 3, hp[3]);
}

__global__ void pool_kernel(const float* __restrict__ h, const int* __restrict__ batch,
                            float* __restrict__ pooled) {
  int idx = blockIdx.x * blockDim.x + threadIdx.x;
  if (idx >= NN * 32) return;
  int n = idx >> 5;
  int c = (idx & 31) << 2;
  int b = batch[n];
  const float* hp = h + (size_t)n * HH + c;
  float* pp = pooled + (size_t)b * HH + c;
  atomicAdd(pp + 0, hp[0]);
  atomicAdd(pp + 1, hp[1]);
  atomicAdd(pp + 2, hp[2]);
  atomicAdd(pp + 3, hp[3]);
}

// coalesced BN partial sums: thread -> channel mapping, 64 rows per block
__global__ void bn_partial_kernel(const float* __restrict__ x, float* __restrict__ part) {
  int c = threadIdx.x & 127;
  int sub = threadIdx.x >> 7;  // 0/1
  int row0 = blockIdx.x * 64;  // grid = NN/64 = 64 blocks
  float s = 0.0f, s2 = 0.0f;
  for (int i = sub; i < 64; i += 2) {
    float v = x[(size_t)(row0 + i) * HH + c];
    s += v;
    s2 += v * v;
  }
  __shared__ float sh[512];
  sh[threadIdx.x] = s;
  sh[256 + threadIdx.x] = s2;
  __syncthreads();
  if (sub == 0) {
    atomicAdd(part + c, sh[c] + sh[128 + c]);
    atomicAdd(part + 128 + c, sh[256 + c] + sh[384 + c]);
  }
}

__global__ void bn_finalize_kernel(const float* __restrict__ part,
                                   const float* __restrict__ gamma,
                                   const float* __restrict__ beta,
                                   float* __restrict__ scl, float* __restrict__ shf) {
  int c = threadIdx.x;  // 128 threads
  float m = part[c] * (1.0f / NN);
  float var = part[128 + c] * (1.0f / NN) - m * m;
  float r = rsqrtf(var + 1e-5f);
  float sc = r * gamma[c];
  scl[c] = sc;
  shf[c] = beta[c] - m * sc;
}

template <bool RELU>
__global__ void bn_apply_kernel(const float* __restrict__ x,
                                const float* __restrict__ scl,
                                const float* __restrict__ shf,
                                float* __restrict__ y) {
  int i = blockIdx.x * blockDim.x + threadIdx.x;
  if (i >= NN * HH) return;
  int c = i & (HH - 1);
  float v = x[i] * scl[c] + shf[c];
  if (RELU) v = fmaxf(v, 0.0f);
  y[i] = v;
}

__global__ void head_out_kernel(const float* __restrict__ r2,
                                const float* __restrict__ Wc,
                                const float* __restrict__ bc,
                                float* __restrict__ out) {
  int i = blockIdx.x * blockDim.x + threadIdx.x;
  if (i >= GG * COUT) return;
  int g = i / COUT, c = i - g * COUT;
  float s = bc[c];
  for (int d = 0; d < HH; ++d) s += r2[g * HH + d] * Wc[d * COUT + c];
  out[i] = s;
}

// ---------------------------------------------------------------------------
// Host-side orchestration
// ---------------------------------------------------------------------------
static void launch_gemm(const float* A, const unsigned* P, const float* bias,
                        const float* res, void* C, int M, int Nc, int K,
                        bool relu, bool out16, hipStream_t s) {
  int waves = (M >> 4) * (Nc >> 6);
  int grid = (waves * 32 + 127) / 128;
  if (out16)
    gemm_bf16_kernel<false, false, true><<<grid, 128, 0, s>>>(A, P, bias, nullptr, C, M, Nc, K);
  else if (relu)
    gemm_bf16_kernel<true, false, false><<<grid, 128, 0, s>>>(A, P, bias, nullptr, C, M, Nc, K);
  else if (res)
    gemm_bf16_kernel<false, true, false><<<grid, 128, 0, s>>>(A, P, bias, res, C, M, Nc, K);
  else
    gemm_bf16_kernel<false, false, false><<<grid, 128, 0, s>>>(A, P, bias, nullptr, C, M, Nc, K);
}

static void launch_pack(const float* W, unsigned* P, int K, int Nc, hipStream_t s) {
  int n = (K >> 1) * Nc;
  pack_weight_kernel<<<(n + 255) / 256, 256, 0, s>>>(W, P, K, Nc);
}

static void launch_bn(const float* x, const float* g, const float* b, float* part,
                      float* scl, float* shf, hipStream_t s) {
  zero_kernel<<<1, 256, 0, s>>>(part, 256);
  bn_partial_kernel<<<NN / 64, 256, 0, s>>>(x, part);
  bn_finalize_kernel<<<1, 128, 0, s>>>(part, g, b, scl, shf);
}

extern "C" void kernel_launch(void* const* d_in, const int* in_sizes, int n_in,
                              void* d_out, int out_size, void* d_ws, size_t ws_size,
                              hipStream_t stream) {
  (void)in_sizes; (void)n_in; (void)out_size; (void)ws_size;
  const float* x        = (const float*)d_in[0];
  const float* lap_pe   = (const float*)d_in[1];
  const float* rw_pe    = (const float*)d_in[2];
  const int*   eidx     = (const int*)d_in[3];
  const int*   batch    = (const int*)d_in[4];
  const float* Wp       = (const float*)d_in[5];
  const float* bp       = (const float*)d_in[6];
  const float* gin_W1   = (const float*)d_in[7];
  const float* gin_b1   = (const float*)d_in[8];
  const float* gin_W2   = (const float*)d_in[9];
  const float* gin_b2   = (const float*)d_in[10];
  const float* bn1_g    = (const float*)d_in[11];
  const float* bn1_b    = (const float*)d_in[12];
  const float* bn2_g    = (const float*)d_in[13];
  const float* bn2_b    = (const float*)d_in[14];
  const float* bn3_g    = (const float*)d_in[15];
  const float* bn3_b    = (const float*)d_in[16];
  const float* attn_in_w  = (const float*)d_in[17];
  const float* attn_in_b  = (const float*)d_in[18];
  const float* attn_out_w = (const float*)d_in[19];
  const float* attn_out_b = (const float*)d_in[20];
  const float* mlp_W1   = (const float*)d_in[21];
  const float* mlp_b1   = (const float*)d_in[22];
  const float* mlp_W2   = (const float*)d_in[23];
  const float* mlp_b2   = (const float*)d_in[24];
  const float* Wr1      = (const float*)d_in[25];
  const float* br1      = (const float*)d_in[26];
  const float* Wr2      = (const float*)d_in[27];
  const float* br2      = (const float*)d_in[28];
  const float* Wc       = (const float*)d_in[29];
  const float* bc       = (const float*)d_in[30];
  float* out = (float*)d_out;

  const int* src = eidx;
  const int* dst = eidx + EE;

  // workspace carve-up (units of 4 bytes, 256B aligned blocks)
  float* ws = (float*)d_ws;
  size_t off = 0;
  auto alloc = [&](size_t n) { float* p = ws + off; off += (n + 63) & ~(size_t)63; return p; };
  float* h    = alloc((size_t)NN * HH);
  float* agg  = alloc((size_t)NN * HH);
  float* ga   = alloc((size_t)NN * HH);   // reused as "out" (g + o_bn)
  float* hid  = alloc((size_t)NN * HH);   // reused as tm (mlp out + res)
  float* tg   = alloc((size_t)NN * HH);   // gin out + res; bn1 in place -> g
  float* ob   = alloc((size_t)NN * HH);
  float* to   = alloc((size_t)NN * HH);   // attn out + res; bn2 in place -> o_bn
  float* mh   = alloc((size_t)NN * 2 * HH);
  __bf16* qkv16 = (__bf16*)alloc((size_t)NN * 3 * HH / 2);   // bf16 qkv
  float* scl  = alloc(HH);
  float* shf  = alloc(HH);
  float* part = alloc(256);
  float* xin  = alloc((size_t)NN * 96);
  float* pooled = alloc((size_t)GG * HH);
  float* r1   = alloc((size_t)GG * HH);
  float* r2   = alloc((size_t)GG * HH);
  // packed bf16 weights (dwords)
  unsigned* wpP   = (unsigned*)alloc(48 * HH);
  unsigned* gw1P  = (unsigned*)alloc(2 * 64 * HH);
  unsigned* gw2P  = (unsigned*)alloc(2 * 64 * HH);
  unsigned* aiP   = (unsigned*)alloc(2 * 64 * 3 * HH);
  unsigned* aoP   = (unsigned*)alloc(2 * 64 * HH);
  unsigned* m1P   = (unsigned*)alloc(2 * 64 * 2 * HH);
  unsigned* m2P   = (unsigned*)alloc((size_t)2 * HH * HH);
  unsigned* r1P   = (unsigned*)alloc(64 * HH);
  unsigned* r2P   = (unsigned*)alloc(64 * HH);

  const int NE = NN * HH;

  // ---- pre-pack all weights to bf16 pair-dwords ----
  pack_wp_kernel<<<(48 * HH + 255) / 256, 256, 0, stream>>>(Wp, wpP);
  for (int l = 0; l < 2; ++l) {
    launch_pack(gin_W1 + (size_t)l * HH * HH, gw1P + (size_t)l * 64 * HH, HH, HH, stream);
    launch_pack(gin_W2 + (size_t)l * HH * HH, gw2P + (size_t)l * 64 * HH, HH, HH, stream);
    launch_pack(attn_in_w + (size_t)l * HH * 3 * HH, aiP + (size_t)l * 64 * 3 * HH, HH, 3 * HH, stream);
    launch_pack(attn_out_w + (size_t)l * HH * HH, aoP + (size_t)l * 64 * HH, HH, HH, stream);
    launch_pack(mlp_W1 + (size_t)l * HH * 2 * HH, m1P + (size_t)l * 64 * 2 * HH, HH, 2 * HH, stream);
    launch_pack(mlp_W2 + (size_t)l * 2 * HH * HH, m2P + (size_t)l * HH * HH, 2 * HH, HH, stream);
  }
  launch_pack(Wr1, r1P, HH, HH, stream);
  launch_pack(Wr2, r2P, HH, HH, stream);

  // ---- input projection: h = concat(x, lap, rw) (padded to 96) @ Wp + bp ----
  build_xin_kernel<<<(NN * 96 + 255) / 256, 256, 0, stream>>>(x, lap_pe, rw_pe, xin);
  launch_gemm(xin, wpP, bp, nullptr, h, NN, HH, 96, false, false, stream);

  for (int l = 0; l < 2; ++l) {
    // ---- GIN conv ----
    zero_kernel<<<(NE + 255) / 256, 256, 0, stream>>>(agg, NE);
    scatter_add_kernel<<<(EE * 32 + 255) / 256, 256, 0, stream>>>(h, src, dst, agg);
    add_kernel<<<(NE + 255) / 256, 256, 0, stream>>>(h, agg, ga, NE);
    launch_gemm(ga, gw1P + (size_t)l * 64 * HH, gin_b1 + l * HH, nullptr, hid,
                NN, HH, HH, true, false, stream);
    launch_gemm(hid, gw2P + (size_t)l * 64 * HH, gin_b2 + l * HH, h, tg,
                NN, HH, HH, false, false, stream);
    launch_bn(tg, bn1_g + l * HH, bn1_b + l * HH, part, scl, shf, stream);
    bn_apply_kernel<false><<<(NE + 255) / 256, 256, 0, stream>>>(tg, scl, shf, tg); // tg = g

    // ---- global attention (qkv produced directly in bf16) ----
    launch_gemm(h, aiP + (size_t)l * 64 * 3 * HH, attn_in_b + l * 3 * HH, nullptr,
                qkv16, NN, 3 * HH, HH, false, true, stream);
    attn_kernel<<<(HEADS * (NN / 16)) / 4, 128, 0, stream>>>(qkv16, ob);
    launch_gemm(ob, aoP + (size_t)l * 64 * HH, attn_out_b + l * HH, h, to,
                NN, HH, HH, false, false, stream);
    launch_bn(to, bn2_g + l * HH, bn2_b + l * HH, part, scl, shf, stream);
    bn_apply_kernel<false><<<(NE + 255) / 256, 256, 0, stream>>>(to, scl, shf, to); // to = o_bn

    // ---- combine + FFN ----
    add_kernel<<<(NE + 255) / 256, 256, 0, stream>>>(tg, to, ga, NE);  // ga = out
    launch_gemm(ga, m1P + (size_t)l * 64 * 2 * HH, mlp_b1 + l * 2 * HH, nullptr,
                mh, NN, 2 * HH, HH, true, false, stream);
    launch_gemm(mh, m2P + (size_t)l * HH * HH, mlp_b2 + l * HH, ga, hid,
                NN, HH, 2 * HH, false, false, stream);
    launch_bn(hid, bn3_g + l * HH, bn3_b + l * HH, part, scl, shf, stream);
    bn_apply_kernel<true><<<(NE + 255) / 256, 256, 0, stream>>>(hid, scl, shf, h); // h = relu(bn3)
  }

  // ---- pooling + MLP head ----
  zero_kernel<<<(GG * HH + 255) / 256, 256, 0, stream>>>(pooled, GG * HH);
  pool_kernel<<<(NN * 32 + 255) / 256, 256, 0, stream>>>(h, batch, pooled);
  launch_gemm(pooled, r1P, br1, nullptr, r1, GG, HH, HH, true, false, stream);
  launch_gemm(r1, r2P, br2, nullptr, r2, GG, HH, HH, true, false, stream);
  head_out_kernel<<<(GG * COUT + 255) / 256, 256, 0, stream>>>(r2, Wc, bc, out);
}
